// SanaMultiscaleLinearAttention_3410204033550
// MI455X (gfx1250) — matrix operand order repaired and checked by reading, not compiled, not run on hardware
//
#include <hip/hip_runtime.h>

#define DI __device__ __forceinline__

typedef __attribute__((ext_vector_type(16))) __bf16 v16bf;
typedef __attribute__((ext_vector_type(2)))  __bf16 v2bf;
typedef __attribute__((ext_vector_type(8)))  float  v8f;

constexpr int HW   = 4096;   // 64*64
constexpr int CIN  = 512;
constexpr int QKVC = 1536;   // 3*inner
constexpr int ATTC = 1024;   // 2*inner
constexpr int NB   = 4;

// native RNE converts (should lower to v_cvt_pk_bf16_f32 / v_cvt_bf16_f32)
DI unsigned pk2bf(float a, float b) {
  v2bf p;
  p[0] = (__bf16)a;
  p[1] = (__bf16)b;
  return __builtin_bit_cast(unsigned, p);
}

union FragAB { v16bf v; unsigned u[8]; };

// gfx1250 async global->LDS copy (16B per lane), ASYNCcnt-tracked
DI void async_copy_b128(unsigned lds_off, const float* gsrc) {
  asm volatile("global_load_async_to_lds_b128 %0, %1, off"
               :: "v"(lds_off), "v"(gsrc) : "memory");
}
DI void wait_asynccnt0() {
  asm volatile("s_wait_asynccnt 0" ::: "memory");
}

// ---------------------------------------------------------------------------
// Kernel 1: qkv[b, ch, hw] = sum_k W[ch%512, k] * x[b, k, hw]   (ch in [0,1536))
// A = weights (M=ch, K=cin) staged as packed bf16; B = x (K=cin, N=hw) staged
// raw f32 via async-LDS copies, converted to bf16 at fragment read.
// grid (256 t-tiles, 24 ch-tiles), block 256 (8 wave32).
// ---------------------------------------------------------------------------
__global__ __launch_bounds__(256) void qkv_gemm_kernel(
    const float* __restrict__ x, const float* __restrict__ wq,
    const float* __restrict__ wk, const float* __restrict__ wv,
    float* __restrict__ qkv)
{
  __shared__ unsigned lds_a[64 * 17];   // bf16 pairs: [ch_local][k/2]
  __shared__ float    lds_b[32 * 65];   // raw f32: [k][t], pad 1 -> banks shift 16/row16

  const int tid  = threadIdx.x;
  const int wave = tid >> 5;
  const int lane = tid & 31;
  const int hi   = lane >> 4;
  const int l16  = lane & 15;
  const int m_sub  = wave & 3;    // 16-row (ch) block
  const int n_half = wave >> 2;   // 32-col (t) block

  const int t_tile  = blockIdx.x * 64;
  const int ch_tile = blockIdx.y * 64;
  const int b   = t_tile >> 12;
  const int hw0 = t_tile & (HW - 1);
  const int proj = ch_tile >> 9;
  const float* __restrict__ w = (proj == 0) ? wq : ((proj == 1) ? wk : wv);
  const int wrow0 = ch_tile & 511;
  const float* __restrict__ xb = x + (size_t)b * CIN * HW;

  const int a_row = tid >> 2;         // 0..63
  const int a_ks  = (tid & 3) * 8;    // 0,8,16,24

  v8f acc0 = {0.f,0.f,0.f,0.f,0.f,0.f,0.f,0.f};
  v8f acc1 = acc0;

  for (int k0 = 0; k0 < CIN; k0 += 32) {
    // Stage A: weight tile [64 ch x 32 k], rows contiguous in K, convert once
    {
      const float* src = w + (size_t)(wrow0 + a_row) * CIN + (k0 + a_ks);
      if (k0 + 32 < CIN) __builtin_prefetch(src + 32, 0, 0);
      float4 f0 = *(const float4*)(src);
      float4 f1 = *(const float4*)(src + 4);
      unsigned* dst = &lds_a[a_row * 17 + (a_ks >> 1)];
      dst[0] = pk2bf(f0.x, f0.y);
      dst[1] = pk2bf(f0.z, f0.w);
      dst[2] = pk2bf(f1.x, f1.y);
      dst[3] = pk2bf(f1.z, f1.w);
    }
    // Stage B: async copy x tile [32 k x 64 t] raw f32 -> LDS (16B per lane)
    #pragma unroll
    for (int hpass = 0; hpass < 2; ++hpass) {
      const int idx = tid + hpass * 256;     // 0..511
      const int kk  = idx >> 4;              // 0..31
      const int t4  = (idx & 15) * 4;        // 0..60
      async_copy_b128((unsigned)(unsigned long long)&lds_b[kk * 65 + t4],
                      xb + (size_t)(k0 + kk) * HW + hw0 + t4);
    }
    wait_asynccnt0();
    __syncthreads();

    FragAB fa, fb0, fb1;
    {
      // A frag: lane<16 row m, K 0..15; lane>=16 same rows, K 16..31
      const unsigned* pa = &lds_a[(m_sub * 16 + l16) * 17 + hi * 8];
      #pragma unroll
      for (int r = 0; r < 8; ++r) { fa.u[r] = pa[r]; }
      // B frag: lane<16 col t, K 0..15; lane>=16 same cols, K 16..31
      const float* pb = &lds_b[(hi * 16) * 65 + n_half * 32 + l16];
      #pragma unroll
      for (int r = 0; r < 8; ++r) {
        fb0.u[r] = pk2bf(pb[(2 * r) * 65],      pb[(2 * r + 1) * 65]);
        fb1.u[r] = pk2bf(pb[(2 * r) * 65 + 16], pb[(2 * r + 1) * 65 + 16]);
      }
    }
    acc0 = __builtin_amdgcn_wmma_f32_16x16x32_bf16(false, fa.v, false, fb0.v,
                                                   (short)0, acc0, false, false);
    acc1 = __builtin_amdgcn_wmma_f32_16x16x32_bf16(false, fa.v, false, fb1.v,
                                                   (short)0, acc1, false, false);
    __syncthreads();
  }

  // C layout: VGPR v, lane L -> row m = v + 8*(L>=16), col = L%16
  const int ch0  = ch_tile + m_sub * 16 + hi * 8;
  const int hw_a = hw0 + n_half * 32 + l16;
  float* o = qkv + ((size_t)b * QKVC + ch0) * HW + hw_a;
  #pragma unroll
  for (int v = 0; v < 8; ++v) {
    o[(size_t)v * HW]      = acc0[v];
    o[(size_t)v * HW + 16] = acc1[v];
  }
}

// ---------------------------------------------------------------------------
// Kernel 2: ms = grouped_pw(dwconv5x5_circular(qkv)); 192 groups x 8 ch.
// grid (4 spatial tiles, 192 groups, 4 batch), block 256; 32x32 tile + halo in LDS.
// ---------------------------------------------------------------------------
__global__ __launch_bounds__(256) void ms_conv_kernel(
    const float* __restrict__ qkv, const float* __restrict__ dw_w,
    const float* __restrict__ pw_w, float* __restrict__ ms)
{
  __shared__ float tile[8][36 * 37];
  __shared__ float wdw[8 * 25];
  __shared__ float wpw[64];
  const int tid = threadIdx.x;
  const int tx = blockIdx.x & 1, ty = blockIdx.x >> 1;
  const int g = blockIdx.y;
  const int b = blockIdx.z;

  const float* src = qkv + ((size_t)b * QKVC + (size_t)g * 8) * HW;
  for (int idx = tid; idx < 8 * 36 * 36; idx += 256) {
    const int ci = idx / (36 * 36);
    const int r  = idx - ci * 36 * 36;
    const int y  = r / 36;
    const int xx = r - y * 36;
    const int gy = (ty * 32 + y - 2) & 63;   // circular wrap
    const int gx = (tx * 32 + xx - 2) & 63;
    tile[ci][y * 37 + xx] = src[(size_t)ci * HW + gy * 64 + gx];
  }
  if (tid < 200) wdw[tid] = dw_w[(size_t)g * 200 + tid];
  if (tid < 64)  wpw[tid] = pw_w[(size_t)g * 64 + tid];
  __syncthreads();

  float* dst = ms + ((size_t)b * QKVC + (size_t)g * 8) * HW;
  for (int p = tid; p < 1024; p += 256) {
    const int y = p >> 5, xx = p & 31;
    float dwv[8];
    #pragma unroll
    for (int ci = 0; ci < 8; ++ci) {
      float s = 0.f;
      #pragma unroll
      for (int dy = 0; dy < 5; ++dy)
        #pragma unroll
        for (int dx = 0; dx < 5; ++dx)
          s += tile[ci][(y + dy) * 37 + (xx + dx)] * wdw[ci * 25 + dy * 5 + dx];
      dwv[ci] = s;
    }
    const int ghw = (ty * 32 + y) * 64 + tx * 32 + xx;
    #pragma unroll
    for (int co = 0; co < 8; ++co) {
      float a = 0.f;
      #pragma unroll
      for (int ci = 0; ci < 8; ++ci) a += wpw[co * 8 + ci] * dwv[ci];
      dst[(size_t)co * HW + ghw] = a;
    }
  }
}

// ---------------------------------------------------------------------------
// Kernel 3: linear attention per (b, head). scores[9][8] = sum_s v_i*relu(k_j),
// then out_i = (scores[i]·relu(q)) / (scores[8]·relu(q) + eps).
// Heads n<64 live in qkv, n>=64 in ms (24*64 == 1536). grid 512, block 256.
// ---------------------------------------------------------------------------
__global__ __launch_bounds__(256) void attn_kernel(
    const float* __restrict__ qkv, const float* __restrict__ ms,
    float* __restrict__ attn)
{
  __shared__ float swave[8][72];
  __shared__ float sfin[72];
  const int tid = threadIdx.x;
  const int wave = tid >> 5, lane = tid & 31;
  const int b = blockIdx.x >> 7;
  const int n = blockIdx.x & 127;

  const float* base = (n < 64)
      ? qkv + ((size_t)b * QKVC + (size_t)24 * n) * HW
      : ms  + ((size_t)b * QKVC + (size_t)24 * (n - 64)) * HW;
  const float* qp = base;
  const float* kp = base + (size_t)8  * HW;
  const float* vp = base + (size_t)16 * HW;

  float acc[72];
  #pragma unroll
  for (int i = 0; i < 72; ++i) acc[i] = 0.f;

  for (int s = tid; s < HW; s += 256) {
    float kv[8], vv[8];
    #pragma unroll
    for (int j = 0; j < 8; ++j) kv[j] = fmaxf(kp[(size_t)j * HW + s], 0.f);
    #pragma unroll
    for (int i = 0; i < 8; ++i) vv[i] = vp[(size_t)i * HW + s];
    #pragma unroll
    for (int i = 0; i < 8; ++i)
      #pragma unroll
      for (int j = 0; j < 8; ++j)
        acc[i * 8 + j] += vv[i] * kv[j];
    #pragma unroll
    for (int j = 0; j < 8; ++j) acc[64 + j] += kv[j];   // ones-padded v row
  }
  // wave32 tree reduce, then cross-wave via LDS
  #pragma unroll
  for (int q = 0; q < 72; ++q) {
    float v = acc[q];
    v += __shfl_down(v, 16, 32);
    v += __shfl_down(v, 8, 32);
    v += __shfl_down(v, 4, 32);
    v += __shfl_down(v, 2, 32);
    v += __shfl_down(v, 1, 32);
    if (lane == 0) swave[wave][q] = v;
  }
  __syncthreads();
  if (tid < 72) {
    float s = 0.f;
    #pragma unroll
    for (int w = 0; w < 8; ++w) s += swave[w][tid];
    sfin[tid] = s;
  }
  __syncthreads();

  float S[72];
  #pragma unroll
  for (int q = 0; q < 72; ++q) S[q] = sfin[q];

  float* op = attn + ((size_t)b * ATTC + (size_t)n * 8) * HW;
  for (int s = tid; s < HW; s += 256) {
    float qv[8];
    #pragma unroll
    for (int j = 0; j < 8; ++j) qv[j] = fmaxf(qp[(size_t)j * HW + s], 0.f);
    float den = 0.f;
    #pragma unroll
    for (int j = 0; j < 8; ++j) den += S[64 + j] * qv[j];
    const float inv = 1.f / (den + 1e-15f);
    #pragma unroll
    for (int i = 0; i < 8; ++i) {
      float o = 0.f;
      #pragma unroll
      for (int j = 0; j < 8; ++j) o += S[i * 8 + j] * qv[j];
      op[(size_t)i * HW + s] = o * inv;
    }
  }
}

// ---------------------------------------------------------------------------
// Kernel 4: out[b,c,hw] = BN(sum_k w_out[c,k]*attn[b,k,hw]) + x[b,c,hw]
// Same WMMA tiling as kernel 1; K = 1024. grid (256, 8), block 256.
// ---------------------------------------------------------------------------
__global__ __launch_bounds__(256) void out_proj_kernel(
    const float* __restrict__ attn, const float* __restrict__ w_out,
    const float* __restrict__ x,
    const float* __restrict__ bn_w, const float* __restrict__ bn_b,
    const float* __restrict__ bn_mean, const float* __restrict__ bn_var,
    float* __restrict__ out)
{
  __shared__ unsigned lds_a[64 * 17];
  __shared__ float    lds_b[32 * 65];

  const int tid  = threadIdx.x;
  const int wave = tid >> 5;
  const int lane = tid & 31;
  const int hi   = lane >> 4;
  const int l16  = lane & 15;
  const int m_sub  = wave & 3;
  const int n_half = wave >> 2;

  const int t_tile  = blockIdx.x * 64;
  const int ch_tile = blockIdx.y * 64;
  const int b   = t_tile >> 12;
  const int hw0 = t_tile & (HW - 1);
  const float* __restrict__ ab = attn + (size_t)b * ATTC * HW;

  const int a_row = tid >> 2;
  const int a_ks  = (tid & 3) * 8;

  v8f acc0 = {0.f,0.f,0.f,0.f,0.f,0.f,0.f,0.f};
  v8f acc1 = acc0;

  for (int k0 = 0; k0 < ATTC; k0 += 32) {
    {
      const float* src = w_out + (size_t)(ch_tile + a_row) * ATTC + (k0 + a_ks);
      if (k0 + 32 < ATTC) __builtin_prefetch(src + 32, 0, 0);
      float4 f0 = *(const float4*)(src);
      float4 f1 = *(const float4*)(src + 4);
      unsigned* dst = &lds_a[a_row * 17 + (a_ks >> 1)];
      dst[0] = pk2bf(f0.x, f0.y);
      dst[1] = pk2bf(f0.z, f0.w);
      dst[2] = pk2bf(f1.x, f1.y);
      dst[3] = pk2bf(f1.z, f1.w);
    }
    #pragma unroll
    for (int hpass = 0; hpass < 2; ++hpass) {
      const int idx = tid + hpass * 256;
      const int kk  = idx >> 4;
      const int t4  = (idx & 15) * 4;
      async_copy_b128((unsigned)(unsigned long long)&lds_b[kk * 65 + t4],
                      ab + (size_t)(k0 + kk) * HW + hw0 + t4);
    }
    wait_asynccnt0();
    __syncthreads();

    FragAB fa, fb0, fb1;
    {
      const unsigned* pa = &lds_a[(m_sub * 16 + l16) * 17 + hi * 8];
      #pragma unroll
      for (int r = 0; r < 8; ++r) { fa.u[r] = pa[r]; }
      const float* pb = &lds_b[(hi * 16) * 65 + n_half * 32 + l16];
      #pragma unroll
      for (int r = 0; r < 8; ++r) {
        fb0.u[r] = pk2bf(pb[(2 * r) * 65],      pb[(2 * r + 1) * 65]);
        fb1.u[r] = pk2bf(pb[(2 * r) * 65 + 16], pb[(2 * r + 1) * 65 + 16]);
      }
    }
    acc0 = __builtin_amdgcn_wmma_f32_16x16x32_bf16(false, fa.v, false, fb0.v,
                                                   (short)0, acc0, false, false);
    acc1 = __builtin_amdgcn_wmma_f32_16x16x32_bf16(false, fa.v, false, fb1.v,
                                                   (short)0, acc1, false, false);
    __syncthreads();
  }

  const int ch0  = ch_tile + m_sub * 16 + hi * 8;
  const int hw_a = hw0 + n_half * 32 + l16;
  const float* xr = x   + ((size_t)b * CIN + ch0) * HW + hw_a;
  float*       o  = out + ((size_t)b * CIN + ch0) * HW + hw_a;
  #pragma unroll
  for (int v = 0; v < 8; ++v) {
    const int c    = ch0 + v;
    const float sc = bn_w[c] * rsqrtf(bn_var[c] + 1e-5f);
    const float mu = bn_mean[c];
    const float bo = bn_b[c];
    o[(size_t)v * HW]      = (acc0[v] - mu) * sc + bo + xr[(size_t)v * HW];
    o[(size_t)v * HW + 16] = (acc1[v] - mu) * sc + bo + xr[(size_t)v * HW + 16];
  }
}

// ---------------------------------------------------------------------------
extern "C" void kernel_launch(void* const* d_in, const int* in_sizes, int n_in,
                              void* d_out, int out_size, void* d_ws, size_t ws_size,
                              hipStream_t stream) {
  const float* x     = (const float*)d_in[0];
  const float* wq    = (const float*)d_in[1];
  const float* wk    = (const float*)d_in[2];
  const float* wv    = (const float*)d_in[3];
  const float* dw_w  = (const float*)d_in[4];
  const float* pw_w  = (const float*)d_in[5];
  const float* w_out = (const float*)d_in[6];
  const float* bn_w  = (const float*)d_in[7];
  const float* bn_b  = (const float*)d_in[8];
  const float* bn_m  = (const float*)d_in[9];
  const float* bn_v  = (const float*)d_in[10];
  float* out = (float*)d_out;

  float* qkv_ws  = (float*)d_ws;                        // 4*1536*4096 f32 (96 MB)
  float* ms_ws   = qkv_ws + (size_t)NB * QKVC * HW;     // 96 MB
  float* attn_ws = ms_ws  + (size_t)NB * QKVC * HW;     // 4*1024*4096 f32 (64 MB)

  qkv_gemm_kernel<<<dim3(256, 24), 256, 0, stream>>>(x, wq, wk, wv, qkv_ws);
  ms_conv_kernel<<<dim3(4, 192, NB), 256, 0, stream>>>(qkv_ws, dw_w, pw_w, ms_ws);
  attn_kernel<<<dim3(512), 256, 0, stream>>>(qkv_ws, ms_ws, attn_ws);
  out_proj_kernel<<<dim3(256, 8), 256, 0, stream>>>(attn_ws, w_out, x,
                                                    bn_w, bn_b, bn_m, bn_v, out);
}
